// SwinTransformerBlock_14877766713539
// MI455X (gfx1250) — matrix operand order repaired and checked by reading, not compiled, not run on hardware
//
#include <hip/hip_runtime.h>
#include <math.h>

typedef _Float16 f16;
typedef __attribute__((ext_vector_type(16))) _Float16 v16h;
typedef __attribute__((ext_vector_type(8)))  float    v8f;

#define BB    8
#define HH    224
#define CC    96
#define WS    7
#define NWIN  49            // tokens per window
#define NWX   32            // windows per row (224/7)
#define NWTOT (BB*NWX*NWX)  // 8192 windows
#define MTOK  401408        // total tokens = 8192*49
#define HEADS 3
#define HDIM  32
#define SHIFTP 3
#define SCALE_Q 0.17677669529663689f  // 32^-0.5
#define EPSLN 1e-5f

// ---------------------------------------------------------------------------
// helpers
// ---------------------------------------------------------------------------
__device__ __forceinline__ float wave_sum32(float v) {
  #pragma unroll
  for (int m = 1; m < 32; m <<= 1) v += __shfl_xor(v, m, 32);
  return v;
}
__device__ __forceinline__ float grp16_max(float v) {
  #pragma unroll
  for (int m = 1; m < 16; m <<= 1) v = fmaxf(v, __shfl_xor(v, m, 32));
  return v;
}
__device__ __forceinline__ float grp16_sum(float v) {
  #pragma unroll
  for (int m = 1; m < 16; m <<= 1) v += __shfl_xor(v, m, 32);
  return v;
}

// Swin shift-mask region id: pos = wcoord*7 + inner; regions [0,217) [217,221) [221,224)
__device__ __forceinline__ int region_id(int wcoord, int inner) {
  int pos = wcoord * WS + inner;
  return (pos >= 221) ? 2 : ((pos >= 217) ? 1 : 0);
}

// A-fragment (16x32 f16, M x K) from LDS row-major [row][k], row stride in halfs
__device__ __forceinline__ v16h load_afrag(const f16* src, int row0, int stride,
                                           int koff, int lm, int hf) {
  v16h a;
  #pragma unroll
  for (int r = 0; r < 8; ++r) {
    int kb = 2 * r + 16 * hf + koff;
    const f16* p = src + (size_t)(row0 + lm) * stride + kb;
    a[2 * r]     = p[0];
    a[2 * r + 1] = p[1];
  }
  return a;
}

// B-fragment (32x16 f16, K x N) from LDS [k][n] with given stride; col0 = tile col
__device__ __forceinline__ v16h load_bfrag(const f16* src, int col0, int stride,
                                           int koff, int lm, int hf) {
  v16h b;
  #pragma unroll
  for (int r = 0; r < 8; ++r) {
    int kk = 2 * r + 16 * hf + koff;
    b[2 * r]     = src[(size_t)kk * stride + col0 + lm];
    b[2 * r + 1] = src[(size_t)(kk + 1) * stride + col0 + lm];
  }
  return b;
}

union U8h { uint4 u; f16 h[8]; };

// ---------------------------------------------------------------------------
// weight transpose + fp32->fp16  : W[N][K] -> Wt[K][N]
// ---------------------------------------------------------------------------
__global__ void conv_transpose_kernel(const float* __restrict__ W,
                                      f16* __restrict__ Wt, int Nrows, int K) {
  int i = blockIdx.x * 256 + threadIdx.x;
  if (i < Nrows * K) {
    int n = i / K, k = i - n * K;
    Wt[(size_t)k * Nrows + n] = (f16)W[i];
  }
}

// ---------------------------------------------------------------------------
// LN1 + cyclic shift + window partition, fp32 -> fp16 (window-ordered tokens)
// ---------------------------------------------------------------------------
__global__ void __launch_bounds__(256)
ln1_shift_window_kernel(const float* __restrict__ x, const float* __restrict__ g,
                        const float* __restrict__ bta, f16* __restrict__ xw) {
  int wave = threadIdx.x >> 5, lane = threadIdx.x & 31;
  int t = blockIdx.x * 8 + wave;            // window-ordered token id
  int win = t / NWIN, n = t - win * NWIN;
  int b   = win >> 10;                      // /1024 windows per image
  int wib = win & 1023;
  int wh = wib >> 5, ww = wib & 31;
  int i = n / WS, j = n - i * WS;
  int hsrc = wh * WS + i + SHIFTP;  if (hsrc >= HH) hsrc -= HH;
  int wsrc = ww * WS + j + SHIFTP;  if (wsrc >= HH) wsrc -= HH;
  size_t base = ((size_t)(b * HH + hsrc) * HH + wsrc) * CC;

  float v0 = x[base + lane], v1 = x[base + lane + 32], v2 = x[base + lane + 64];
  float mu = wave_sum32(v0 + v1 + v2) * (1.0f / CC);
  float d0 = v0 - mu, d1 = v1 - mu, d2 = v2 - mu;
  float var = wave_sum32(d0 * d0 + d1 * d1 + d2 * d2) * (1.0f / CC);
  float rs = rsqrtf(var + EPSLN);
  size_t ob = (size_t)t * CC;
  xw[ob + lane]      = (f16)(d0 * rs * g[lane]      + bta[lane]);
  xw[ob + lane + 32] = (f16)(d1 * rs * g[lane + 32] + bta[lane + 32]);
  xw[ob + lane + 64] = (f16)(d2 * rs * g[lane + 64] + bta[lane + 64]);
}

// ---------------------------------------------------------------------------
// LN2 : X1 (fp32, natural token order) -> Y16 (fp16)
// ---------------------------------------------------------------------------
__global__ void __launch_bounds__(256)
ln2_kernel(const float* __restrict__ x, const float* __restrict__ g,
           const float* __restrict__ bta, f16* __restrict__ y) {
  int wave = threadIdx.x >> 5, lane = threadIdx.x & 31;
  int t = blockIdx.x * 8 + wave;
  size_t base = (size_t)t * CC;
  float v0 = x[base + lane], v1 = x[base + lane + 32], v2 = x[base + lane + 64];
  float mu = wave_sum32(v0 + v1 + v2) * (1.0f / CC);
  float d0 = v0 - mu, d1 = v1 - mu, d2 = v2 - mu;
  float var = wave_sum32(d0 * d0 + d1 * d1 + d2 * d2) * (1.0f / CC);
  float rs = rsqrtf(var + EPSLN);
  y[base + lane]      = (f16)(d0 * rs * g[lane]      + bta[lane]);
  y[base + lane + 32] = (f16)(d1 * rs * g[lane + 32] + bta[lane + 32]);
  y[base + lane + 64] = (f16)(d2 * rs * g[lane + 64] + bta[lane + 64]);
}

// ---------------------------------------------------------------------------
// Generic WMMA GEMM: C[M,N] = A[M,K](f16) * Bt[K,N](f16) + bias, MODE epilogue
// block = 128 threads (4 waves); tile 64(M) x 32(N); K step 32
// MODE 0: qkv -> out16 | 1: proj scatter+resid fp32 | 2: fc1 GELU | 3: fc2 resid fp32
// ---------------------------------------------------------------------------
#define ASTR 40  // LDS row stride in halfs (80B: keeps 8-half columns 16B aligned)

template <int MODE>
__global__ void __launch_bounds__(128)
gemm_wmma_kernel(const f16* __restrict__ A, const f16* __restrict__ Bt,
                 const float* __restrict__ bias, f16* __restrict__ out16,
                 float* __restrict__ out32, const float* __restrict__ resid,
                 int M, int N, int K) {
  __shared__ f16 As[64 * ASTR];
  __shared__ f16 Bs[32 * ASTR];
  int tid = threadIdx.x;
  int wave = tid >> 5, lane = tid & 31;
  int lm = lane & 15, hf = lane >> 4;
  int tileM = blockIdx.x * 64;
  int tileN = blockIdx.y * 32;

  v8f acc[2] = {};

  for (int kb = 0; kb < K; kb += 32) {
    // stage A: 64 rows x 4 uint4 (b128) each
    #pragma unroll
    for (int e = tid; e < 256; e += 128) {
      int r = e >> 2, c8 = (e & 3) * 8;
      *(uint4*)(&As[r * ASTR + c8]) =
          *(const uint4*)(A + (size_t)(tileM + r) * K + kb + c8);
    }
    // stage B: 32 rows x 4 uint4, one per thread
    {
      int r = tid >> 2, c8 = (tid & 3) * 8;
      *(uint4*)(&Bs[r * ASTR + c8]) =
          *(const uint4*)(Bt + (size_t)(kb + r) * N + tileN + c8);
    }
    __syncthreads();

    v16h af = load_afrag(As, 16 * wave, ASTR, 0, lm, hf);
    #pragma unroll
    for (int t = 0; t < 2; ++t) {
      v16h bf = load_bfrag(Bs, t * 16, ASTR, 0, lm, hf);
      acc[t] = __builtin_amdgcn_wmma_f32_16x16x32_f16(
          false, af, false, bf, (short)0, acc[t], false, false);
    }
    __syncthreads();
  }

  #pragma unroll
  for (int t = 0; t < 2; ++t) {
    #pragma unroll
    for (int r = 0; r < 8; ++r) {
      int row = tileM + 16 * wave + r + 8 * hf;
      int col = tileN + t * 16 + lm;
      float v = acc[t][r] + bias[col];
      if (MODE == 0) {
        out16[(size_t)row * N + col] = (f16)v;
      } else if (MODE == 1) {
        int win = row / NWIN, n = row - win * NWIN;
        int b = win >> 10, wib = win & 1023;
        int wh = wib >> 5, ww = wib & 31;
        int i = n / WS, j = n - i * WS;
        int h0 = wh * WS + i + SHIFTP; if (h0 >= HH) h0 -= HH;
        int w0 = ww * WS + j + SHIFTP; if (w0 >= HH) w0 -= HH;
        size_t oi = ((size_t)(b * HH + h0) * HH + w0) * CC + col;
        out32[oi] = resid[oi] + v;
      } else if (MODE == 2) {
        float gel = 0.5f * v * (1.0f + erff(v * 0.7071067811865476f));
        out16[(size_t)row * N + col] = (f16)gel;
      } else {
        size_t oi = (size_t)row * N + col;
        out32[oi] = resid[oi] + v;
      }
    }
  }
}

// ---------------------------------------------------------------------------
// Windowed attention: one block per (window, head); 128 threads = 4 waves.
// S = (Q*scale) K^T + analytic shift-mask ; softmax ; O = P V  (49 padded to 64)
// ---------------------------------------------------------------------------
__global__ void __launch_bounds__(128)
attention_kernel(const f16* __restrict__ qkv, f16* __restrict__ obuf) {
  __shared__ f16 Qs[64 * ASTR];
  __shared__ f16 Ks[64 * ASTR];
  __shared__ f16 Vs[64 * ASTR];
  __shared__ f16 Ps[64 * 66];

  int win = blockIdx.x, hd = blockIdx.y;
  int tid = threadIdx.x;
  int wave = tid >> 5, lane = tid & 31;
  int lm = lane & 15, hf = lane >> 4;
  int wib = win & 1023;
  int wh = wib >> 5, ww = wib & 31;

  // stage Q (scaled), K, V with b128 loads: rows 0..48 valid, 49..63 zero
  #pragma unroll
  for (int e = tid; e < 256; e += 128) {
    int r = e >> 2, c8 = (e & 3) * 8;
    U8h q, k, v;
    if (r < NWIN) {
      const f16* base = qkv + (size_t)(win * NWIN + r) * (3 * CC) + hd * HDIM + c8;
      q.u = *(const uint4*)(base);
      k.u = *(const uint4*)(base + CC);
      v.u = *(const uint4*)(base + 2 * CC);
      #pragma unroll
      for (int z = 0; z < 8; ++z) q.h[z] = q.h[z] * (f16)SCALE_Q;
    } else {
      q.u = uint4{0, 0, 0, 0};
      k.u = uint4{0, 0, 0, 0};
      v.u = uint4{0, 0, 0, 0};
    }
    *(uint4*)(&Qs[r * ASTR + c8]) = q.u;
    *(uint4*)(&Ks[r * ASTR + c8]) = k.u;
    *(uint4*)(&Vs[r * ASTR + c8]) = v.u;
  }
  __syncthreads();

  int row0 = 16 * wave;
  v16h aq = load_afrag(Qs, row0, ASTR, 0, lm, hf);
  v8f sacc[4] = {};
  #pragma unroll
  for (int t = 0; t < 4; ++t) {
    // B[kk][n] = K[t*16+n][kk] : read rows of Ks
    v16h bk;
    #pragma unroll
    for (int r = 0; r < 8; ++r) {
      int kk = 2 * r + 16 * hf;
      const f16* p = Ks + (size_t)(t * 16 + lm) * ASTR + kk;
      bk[2 * r] = p[0];
      bk[2 * r + 1] = p[1];
    }
    sacc[t] = __builtin_amdgcn_wmma_f32_16x16x32_f16(
        false, aq, false, bk, (short)0, sacc[t], false, false);
  }

  // analytic shift-mask region ids for this lane's 4 columns
  int idc[4];
  #pragma unroll
  for (int t = 0; t < 4; ++t) {
    int col = t * 16 + lm;
    int ic = col / WS, jc = col - ic * WS;
    idc[t] = (col < NWIN) ? (region_id(wh, ic) * 3 + region_id(ww, jc)) : -1;
  }

  // mask + softmax per row (C-layout: row = r + 8*hf, col = 16*t + lm)
  #pragma unroll
  for (int r = 0; r < 8; ++r) {
    int row = row0 + r + 8 * hf;
    int ir = row / WS, jr = row - ir * WS;
    int idr = region_id(wh, ir) * 3 + region_id(ww, jr);
    float mx = -1e30f;
    #pragma unroll
    for (int t = 0; t < 4; ++t) {
      float s = sacc[t][r];
      s = (idc[t] == idr) ? s : s - 100.0f;
      s = (idc[t] < 0) ? -1e30f : s;
      sacc[t][r] = s;
      mx = fmaxf(mx, s);
    }
    mx = grp16_max(mx);
    float sum = 0.0f;
    #pragma unroll
    for (int t = 0; t < 4; ++t) {
      float e = __expf(sacc[t][r] - mx);
      sacc[t][r] = e;
      sum += e;
    }
    sum = grp16_sum(sum);
    float inv = 1.0f / sum;
    #pragma unroll
    for (int t = 0; t < 4; ++t)
      Ps[(size_t)row * 66 + t * 16 + lm] = (f16)(sacc[t][r] * inv);
  }
  __syncthreads();

  // O = P(16x64) * V(64x32): 2 k-chunks x 2 n-tiles
  v16h ap0 = load_afrag(Ps, row0, 66, 0, lm, hf);
  v16h ap1 = load_afrag(Ps, row0, 66, 32, lm, hf);
  #pragma unroll
  for (int tn = 0; tn < 2; ++tn) {
    v8f oacc = {};
    v16h bv0 = load_bfrag(Vs, tn * 16, ASTR, 0, lm, hf);
    oacc = __builtin_amdgcn_wmma_f32_16x16x32_f16(
        false, ap0, false, bv0, (short)0, oacc, false, false);
    v16h bv1 = load_bfrag(Vs, tn * 16, ASTR, 32, lm, hf);
    oacc = __builtin_amdgcn_wmma_f32_16x16x32_f16(
        false, ap1, false, bv1, (short)0, oacc, false, false);
    #pragma unroll
    for (int r = 0; r < 8; ++r) {
      int row = row0 + r + 8 * hf;
      if (row < NWIN) {
        obuf[(size_t)(win * NWIN + row) * CC + hd * HDIM + tn * 16 + lm] =
            (f16)oacc[r];
      }
    }
  }
}

// ---------------------------------------------------------------------------
// host driver
// ---------------------------------------------------------------------------
extern "C" void kernel_launch(void* const* d_in, const int* in_sizes, int n_in,
                              void* d_out, int out_size, void* d_ws, size_t ws_size,
                              hipStream_t stream) {
  (void)in_sizes; (void)n_in; (void)out_size; (void)ws_size;
  const float* x       = (const float*)d_in[0];
  // d_in[1] (mask) is reproduced analytically inside attention_kernel
  const float* norm1_g = (const float*)d_in[2];
  const float* norm1_b = (const float*)d_in[3];
  const float* qkv_w   = (const float*)d_in[4];
  const float* qkv_b   = (const float*)d_in[5];
  const float* proj_w  = (const float*)d_in[6];
  const float* proj_b  = (const float*)d_in[7];
  const float* norm2_g = (const float*)d_in[8];
  const float* norm2_b = (const float*)d_in[9];
  const float* fc1_w   = (const float*)d_in[10];
  const float* fc1_b   = (const float*)d_in[11];
  const float* fc2_w   = (const float*)d_in[12];
  const float* fc2_b   = (const float*)d_in[13];
  float* out = (float*)d_out;

  char* ws = (char*)d_ws;
  f16* wqkv  = (f16*)(ws + 0);                 //  96*288 f16
  f16* wproj = (f16*)(ws + 55296);             //  96* 96 f16
  f16* wfc1  = (f16*)(ws + 73728);             //  96*384 f16
  f16* wfc2  = (f16*)(ws + 147456);            // 384* 96 f16
  f16* buf1  = (f16*)(ws + 221184);            // MTOK* 96 f16 (xw / obuf / y16)
  f16* qkvb  = (f16*)(ws + 77291520ULL);       // MTOK*288 f16
  float* X1  = (float*)(ws + 308502528ULL);    // MTOK* 96 f32
  f16* H1    = (f16*)(ws + 462643200ULL);      // MTOK*384 f16

  // 1. weights -> f16, transposed to [K][N]
  conv_transpose_kernel<<<(288 * 96 + 255) / 256, 256, 0, stream>>>(qkv_w, wqkv, 288, 96);
  conv_transpose_kernel<<<(96 * 96 + 255) / 256, 256, 0, stream>>>(proj_w, wproj, 96, 96);
  conv_transpose_kernel<<<(384 * 96 + 255) / 256, 256, 0, stream>>>(fc1_w, wfc1, 384, 96);
  conv_transpose_kernel<<<(96 * 384 + 255) / 256, 256, 0, stream>>>(fc2_w, wfc2, 96, 384);

  // 2. LN1 + shift + window partition -> buf1 (xw, window-ordered f16)
  ln1_shift_window_kernel<<<MTOK / 8, 256, 0, stream>>>(x, norm1_g, norm1_b, buf1);

  // 3. QKV GEMM: (MTOK x 96) * (96 x 288) -> qkvb f16
  gemm_wmma_kernel<0><<<dim3(MTOK / 64, 288 / 32), 128, 0, stream>>>(
      buf1, wqkv, qkv_b, qkvb, nullptr, nullptr, MTOK, 288, 96);

  // 4. windowed attention -> buf1 (obuf, window-ordered f16)
  attention_kernel<<<dim3(NWTOT, HEADS), 128, 0, stream>>>(qkvb, buf1);

  // 5. proj GEMM + window-reverse + unshift + residual -> X1 (f32, natural order)
  gemm_wmma_kernel<1><<<dim3(MTOK / 64, 96 / 32), 128, 0, stream>>>(
      buf1, wproj, proj_b, nullptr, X1, x, MTOK, 96, 96);

  // 6. LN2 -> buf1 (y16)
  ln2_kernel<<<MTOK / 8, 256, 0, stream>>>(X1, norm2_g, norm2_b, buf1);

  // 7. fc1 GEMM + exact GELU -> H1 f16
  gemm_wmma_kernel<2><<<dim3(MTOK / 64, 384 / 32), 128, 0, stream>>>(
      buf1, wfc1, fc1_b, H1, nullptr, nullptr, MTOK, 384, 96);

  // 8. fc2 GEMM + residual(X1) -> out (f32)
  gemm_wmma_kernel<3><<<dim3(MTOK / 64, 96 / 32), 128, 0, stream>>>(
      H1, wfc2, fc2_b, nullptr, out, X1, MTOK, 96, 384);
}
